// TemporalAttention_60172491817305
// MI455X (gfx1250) — matrix operand order repaired and checked by reading, not compiled, not run on hardware
//
#include <hip/hip_runtime.h>
#include <hip/hip_bf16.h>
#include <math.h>

// ---------------- problem constants ----------------
#define B_   4
#define S_   4096
#define D_   1024
#define DK_  128
#define TP_  16
#define TE_  64

typedef __bf16 bf16_t;
typedef bf16_t v16bf __attribute__((ext_vector_type(16)));
typedef bf16_t v8bf  __attribute__((ext_vector_type(8)));
typedef bf16_t v4bf  __attribute__((ext_vector_type(4)));
typedef float  v8f   __attribute__((ext_vector_type(8)));

union BF16x16 { v16bf v; v8bf h[2]; };

__device__ __forceinline__ v8f v8f_zero() {
  v8f z = {0.f,0.f,0.f,0.f,0.f,0.f,0.f,0.f};
  return z;
}

// D = A(16x32 bf16) * B(32x16 bf16) + C(16x16 f32)
__device__ __forceinline__ v8f wmma_bf16(v16bf a, v16bf b, v8f c) {
  return __builtin_amdgcn_wmma_f32_16x16x32_bf16(
      /*neg_a=*/false, a, /*neg_b=*/false, b,
      /*c_mod=*/(short)0, c, /*reuse_a=*/false, /*reuse_b=*/false);
}

// A fragment (16x32, bf16) from row-major [M x K] source.
// lanes 0-15: v0..3 = K k..k+7, v4..7 = K k+16..k+23 (row = r+lane)
// lanes16-31: v0..3 = K k+8..k+15, v4..7 = K k+24..k+31
__device__ __forceinline__ v16bf load_frag_a(const bf16_t* p, int ld, int r, int k, int lane) {
  BF16x16 u;
  const bf16_t* row = p + (size_t)(r + (lane & 15)) * ld;
  int k0 = k + ((lane >> 4) << 3);
  u.h[0] = *(const v8bf*)(row + k0);
  u.h[1] = *(const v8bf*)(row + k0 + 16);
  return u.v;
}

// B fragment (32x16, bf16) from row-major [N x K] source (i.e. transposed operand).
// lanes 0-15: v0..3 = K k..k+7, v4..7 = K k+16..k+23 ; lanes16-31: v0..3 = K k+8..k+15, v4..7 = K k+24..k+31 (row = n+lane)
__device__ __forceinline__ v16bf load_frag_b(const bf16_t* p, int ld, int n, int k, int lane) {
  BF16x16 u;
  const bf16_t* row = p + (size_t)(n + (lane & 15)) * ld;
  int k0 = k + ((lane >> 4) << 3);
  u.h[0] = *(const v8bf*)(row + k0);
  u.h[1] = *(const v8bf*)(row + k0 + 16);
  return u.v;
}

// ---------------- K0: x fp32 -> bf16 ----------------
__global__ __launch_bounds__(256) void cvt_x_kernel(const float* __restrict__ x,
                                                    bf16_t* __restrict__ xb) {
  size_t idx = ((size_t)blockIdx.x * 256 + threadIdx.x) * 4;
  float4 v = *(const float4*)(x + idx);
  v4bf o;
  o[0] = (bf16_t)v.x; o[1] = (bf16_t)v.y; o[2] = (bf16_t)v.z; o[3] = (bf16_t)v.w;
  *(v4bf*)(xb + idx) = o;
}

// ---------------- K1: E, T_scaled, bq' (one block per batch) ----------------
__global__ __launch_bounds__(128) void prep_kernel(const float* __restrict__ tpnts,
                                                   const float* __restrict__ Wt,
                                                   const float* __restrict__ bt,
                                                   const float* __restrict__ bq,
                                                   float* __restrict__ Ts,
                                                   float* __restrict__ bqp) {
  __shared__ float E[TP_][DK_];
  __shared__ float red[128];
  int b = blockIdx.x, t = threadIdx.x;
  for (int tp = 0; tp < TP_; ++tp) {
    float acc = bt[t];
    const float* row = tpnts + ((size_t)b * TP_ + tp) * TE_;
    #pragma unroll 4
    for (int te = 0; te < TE_; ++te) acc += row[te] * Wt[te * DK_ + t];
    E[tp][t] = acc;
  }
  __syncthreads();
  float loc = 0.f;
  for (int tp = 0; tp < TP_; ++tp) loc += E[tp][t] * E[tp][t];
  red[t] = loc;
  __syncthreads();
  for (int off = 64; off > 0; off >>= 1) {
    if (t < off) red[t] += red[t + off];
    __syncthreads();
  }
  float denom = sqrtf((float)S_ * red[0]) + 1e-8f;
  float kf = (float)S_ / denom;
  float* Tsb = Ts + (size_t)b * DK_ * DK_;
  for (int d = 0; d < DK_; ++d) {
    float acc = 0.f;
    #pragma unroll
    for (int tp = 0; tp < TP_; ++tp) acc += E[tp][d] * E[tp][t];
    Tsb[d * DK_ + t] = acc * kf;
  }
  // bq' = bq @ T_scaled  computed from E directly: kf * sum_tp (bq.E[tp]) E[tp][t]
  float c[TP_];
  for (int tp = 0; tp < TP_; ++tp) {
    float a = 0.f;
    for (int d = 0; d < DK_; ++d) a += bq[d] * E[tp][d];
    c[tp] = a;
  }
  float acc = 0.f;
  #pragma unroll
  for (int tp = 0; tp < TP_; ++tp) acc += c[tp] * E[tp][t];
  bqp[b * DK_ + t] = acc * kf;
}

// ---------------- K2a: Wq' ^T = (Wq @ T_scaled)^T per batch, bf16 ----------------
__global__ __launch_bounds__(256) void wqp_kernel(const float* __restrict__ Wq,
                                                  const float* __restrict__ Ts,
                                                  bf16_t* __restrict__ WqpT) {
  int tid = blockIdx.x * 256 + threadIdx.x;   // B*128*1024 threads
  int k = tid & (D_ - 1);
  int n = (tid >> 10) & (DK_ - 1);
  int b = tid >> 17;
  const float* Tsb = Ts + (size_t)b * DK_ * DK_;
  const float* wrow = Wq + (size_t)k * DK_;
  float acc = 0.f;
  #pragma unroll 4
  for (int d = 0; d < DK_; ++d) acc += wrow[d] * Tsb[d * DK_ + n];
  WqpT[((size_t)b * DK_ + n) * D_ + k] = (bf16_t)acc;
}

// ---------------- K2b: W^T bf16 (for Wk, Wv) ----------------
__global__ __launch_bounds__(256) void wtc_kernel(const float* __restrict__ W,
                                                  bf16_t* __restrict__ WT) {
  int tid = blockIdx.x * 256 + threadIdx.x;   // 128*1024 threads
  int k = tid & (D_ - 1);
  int n = tid >> 10;
  WT[(size_t)n * D_ + k] = (bf16_t)W[(size_t)k * DK_ + n];
}

// ---------------- K3: fused QT/K/V projection GEMM (WMMA bf16) ----------------
// grid (S/128, 3, B), 256 threads = 8 waves in 4x2 layout, wave tile 32x64.
__global__ __launch_bounds__(256) void proj_kernel(const bf16_t* __restrict__ xb,
                                                   const bf16_t* __restrict__ WqpT,
                                                   const bf16_t* __restrict__ WkT,
                                                   const bf16_t* __restrict__ WvT,
                                                   const float* __restrict__ bqp,
                                                   const float* __restrict__ bk,
                                                   const float* __restrict__ bv,
                                                   bf16_t* __restrict__ QT,
                                                   bf16_t* __restrict__ Kb,
                                                   bf16_t* __restrict__ Vb) {
  int b = blockIdx.z, y = blockIdx.y;
  const bf16_t* WT;
  const float* bias;
  bf16_t* dst;
  if (y == 0)      { WT = WqpT + (size_t)b * DK_ * D_; bias = bqp + b * DK_; dst = QT; }
  else if (y == 1) { WT = WkT; bias = bk; dst = Kb; }
  else             { WT = WvT; bias = bv; dst = Vb; }
  dst += (size_t)b * S_ * DK_;
  const bf16_t* xbb = xb + (size_t)b * S_ * D_;

  int lane = threadIdx.x & 31, w = threadIdx.x >> 5;
  int m0 = blockIdx.x * 128 + (w >> 1) * 32;
  int n0 = (w & 1) * 64;

  v8f acc[2][4];
  #pragma unroll
  for (int fa = 0; fa < 2; ++fa)
    #pragma unroll
    for (int fn = 0; fn < 4; ++fn) acc[fa][fn] = v8f_zero();

  for (int ks = 0; ks < D_ / 32; ++ks) {
    v16bf a0 = load_frag_a(xbb, D_, m0,      ks * 32, lane);
    v16bf a1 = load_frag_a(xbb, D_, m0 + 16, ks * 32, lane);
    #pragma unroll
    for (int fn = 0; fn < 4; ++fn) {
      v16bf bw = load_frag_b(WT, D_, n0 + 16 * fn, ks * 32, lane);
      acc[0][fn] = wmma_bf16(a0, bw, acc[0][fn]);
      acc[1][fn] = wmma_bf16(a1, bw, acc[1][fn]);
    }
  }
  int lo = lane & 15, hi = lane >> 4;
  #pragma unroll
  for (int fa = 0; fa < 2; ++fa)
    #pragma unroll
    for (int fn = 0; fn < 4; ++fn) {
      int col = n0 + 16 * fn + lo;
      float bb = bias[col];
      #pragma unroll
      for (int i = 0; i < 8; ++i)
        dst[(size_t)(m0 + 16 * fa + i + 8 * hi) * DK_ + col] = (bf16_t)(acc[fa][fn][i] + bb);
    }
}

// ---------------- K4: V -> V^T (bf16, tiled LDS transpose) ----------------
__global__ __launch_bounds__(256) void vtrans_kernel(const bf16_t* __restrict__ Vb,
                                                     bf16_t* __restrict__ Vt) {
  __shared__ bf16_t tile[32][33];
  int b = blockIdx.z;
  int s0 = blockIdx.x * 32, d0 = blockIdx.y * 32;
  int tx = threadIdx.x & 31, ty = threadIdx.x >> 5;
  const bf16_t* src = Vb + (size_t)b * S_ * DK_;
  bf16_t* dstp = Vt + (size_t)b * DK_ * S_;
  #pragma unroll
  for (int i = 0; i < 4; ++i) {
    int r = ty + 8 * i;
    tile[r][tx] = src[(size_t)(s0 + r) * DK_ + d0 + tx];
  }
  __syncthreads();
  #pragma unroll
  for (int i = 0; i < 4; ++i) {
    int r = ty + 8 * i;
    dstp[(size_t)(d0 + r) * S_ + s0 + tx] = tile[tx][r];
  }
}

// ---------------- K5: flash attention, writes raw scaled scores + out ----------------
// grid (S/64, B), 128 threads = 4 waves; each wave owns 16 complete rows.
__global__ __launch_bounds__(128) void attn_kernel(const bf16_t* __restrict__ QT,
                                                   const bf16_t* __restrict__ Kb,
                                                   const bf16_t* __restrict__ Vt,
                                                   float* __restrict__ out,
                                                   float* __restrict__ attn,
                                                   float* __restrict__ stats) {
  __shared__ bf16_t Plds[4][16][136];   // per-wave P tile, padded rows (no barriers needed)
  const int lane = threadIdx.x & 31;
  const int w = threadIdx.x >> 5;
  const int b = blockIdx.y;
  const int m0 = blockIdx.x * 64 + w * 16;
  const int lo = lane & 15, hi = lane >> 4;

  const bf16_t* QTb = QT + (size_t)b * S_ * DK_;
  const bf16_t* Kbb = Kb + (size_t)b * S_ * DK_;
  const bf16_t* Vtb = Vt + (size_t)b * DK_ * S_;
  float* attnb = attn + (size_t)b * S_ * S_;

  v16bf aq[4];
  #pragma unroll
  for (int ks = 0; ks < 4; ++ks) aq[ks] = load_frag_a(QTb, DK_, m0, ks * 32, lane);

  v8f o[8];
  #pragma unroll
  for (int f = 0; f < 8; ++f) o[f] = v8f_zero();
  float m_r[8], s_r[8];
  #pragma unroll
  for (int i = 0; i < 8; ++i) { m_r[i] = -3.0e38f; s_r[i] = 0.f; }

  const float scale = 0.08838834764831845f;   // 1/sqrt(128)

  for (int j = 0; j < S_ / 128; ++j) {
    const int n0 = j * 128;
    v8f sc[8];
    #pragma unroll
    for (int f = 0; f < 8; ++f) sc[f] = v8f_zero();

    // scores 16x128 = QT(16x128) @ K^T
    #pragma unroll
    for (int ks = 0; ks < 4; ++ks)
      #pragma unroll
      for (int f = 0; f < 8; ++f) {
        v16bf bk16 = load_frag_b(Kbb, DK_, n0 + 16 * f, ks * 32, lane);
        sc[f] = wmma_bf16(aq[ks], bk16, sc[f]);
      }

    // scale, local row-max, write raw scaled scores into attn output region
    float tmax[8];
    #pragma unroll
    for (int i = 0; i < 8; ++i) tmax[i] = -3.0e38f;
    #pragma unroll
    for (int f = 0; f < 8; ++f)
      #pragma unroll
      for (int i = 0; i < 8; ++i) {
        float v = sc[f][i] * scale;
        sc[f][i] = v;
        tmax[i] = fmaxf(tmax[i], v);
      }
    #pragma unroll
    for (int i = 0; i < 8; ++i) {
      float* dstp = attnb + (size_t)(m0 + i + 8 * hi) * S_ + n0 + lo;
      #pragma unroll
      for (int f = 0; f < 8; ++f) dstp[16 * f] = sc[f][i];
    }
    // 16-lane row-max reduction (rows are confined to each half-wave)
    #pragma unroll
    for (int i = 0; i < 8; ++i)
      #pragma unroll
      for (int off = 1; off < 16; off <<= 1)
        tmax[i] = fmaxf(tmax[i], __shfl_xor(tmax[i], off, 32));

    // online softmax update
    float alpha[8];
    #pragma unroll
    for (int i = 0; i < 8; ++i) {
      float mn = fmaxf(m_r[i], tmax[i]);
      alpha[i] = __expf(m_r[i] - mn);
      m_r[i] = mn;
      s_r[i] *= alpha[i];
    }
    #pragma unroll
    for (int f = 0; f < 8; ++f)
      #pragma unroll
      for (int i = 0; i < 8; ++i) o[f][i] *= alpha[i];

    float tsum[8];
    #pragma unroll
    for (int i = 0; i < 8; ++i) tsum[i] = 0.f;
    #pragma unroll
    for (int f = 0; f < 8; ++f)
      #pragma unroll
      for (int i = 0; i < 8; ++i) {
        float p = __expf(sc[f][i] - m_r[i]);
        sc[f][i] = p;
        tsum[i] += p;
      }
    #pragma unroll
    for (int i = 0; i < 8; ++i) {
      #pragma unroll
      for (int off = 1; off < 16; off <<= 1) tsum[i] += __shfl_xor(tsum[i], off, 32);
      s_r[i] += tsum[i];
    }

    // P (bf16) -> LDS (per-wave region; same-wave DS ops are in order)
    #pragma unroll
    for (int f = 0; f < 8; ++f)
      #pragma unroll
      for (int i = 0; i < 8; ++i)
        Plds[w][i + 8 * hi][16 * f + lo] = (bf16_t)sc[f][i];

    // O += P(16x128) @ V(128x128): A from LDS, B from V^T row-major [dk x S]
    #pragma unroll
    for (int ks = 0; ks < 4; ++ks) {
      v16bf ap = load_frag_a(&Plds[w][0][0], 136, 0, ks * 32, lane);
      #pragma unroll
      for (int fd = 0; fd < 8; ++fd) {
        v16bf bv16 = load_frag_b(Vtb, S_, 16 * fd, n0 + ks * 32, lane);
        o[fd] = wmma_bf16(ap, bv16, o[fd]);
      }
    }
  }

  // epilogue: out = O / s ; save (m, s) for the normalize pass
  float inv[8];
  #pragma unroll
  for (int i = 0; i < 8; ++i) inv[i] = 1.0f / s_r[i];
  float* outb = out + (size_t)b * S_ * DK_;
  #pragma unroll
  for (int fd = 0; fd < 8; ++fd)
    #pragma unroll
    for (int i = 0; i < 8; ++i)
      outb[(size_t)(m0 + i + 8 * hi) * DK_ + 16 * fd + lo] = o[fd][i] * inv[i];

  if (lo == 0) {
    #pragma unroll
    for (int i = 0; i < 8; ++i) {
      size_t r = (size_t)b * S_ + m0 + i + 8 * hi;
      stats[2 * r] = m_r[i];
      stats[2 * r + 1] = s_r[i];
    }
  }
}

// ---------------- K6: attn = exp(raw - m) / s  (in place) ----------------
__global__ __launch_bounds__(256) void norm_kernel(float* __restrict__ attn,
                                                   const float* __restrict__ stats) {
  size_t idx = ((size_t)blockIdx.x * 256 + threadIdx.x) * 4;
  size_t r = idx / S_;                  // global (b*S + si) row, same for all 4 elems
  float m = stats[2 * r];
  float invs = 1.0f / stats[2 * r + 1];
  float4 v = *(float4*)(attn + idx);
  v.x = __expf(v.x - m) * invs;
  v.y = __expf(v.y - m) * invs;
  v.z = __expf(v.z - m) * invs;
  v.w = __expf(v.w - m) * invs;
  *(float4*)(attn + idx) = v;
}

// ---------------- host launcher ----------------
extern "C" void kernel_launch(void* const* d_in, const int* in_sizes, int n_in,
                              void* d_out, int out_size, void* d_ws, size_t ws_size,
                              hipStream_t stream) {
  const float* x     = (const float*)d_in[0];
  const float* tpnts = (const float*)d_in[1];
  const float* Wq    = (const float*)d_in[2];
  const float* bq    = (const float*)d_in[3];
  const float* Wk    = (const float*)d_in[4];
  const float* bk    = (const float*)d_in[5];
  const float* Wv    = (const float*)d_in[6];
  const float* bv    = (const float*)d_in[7];
  const float* Wt    = (const float*)d_in[8];
  const float* bt    = (const float*)d_in[9];

  float* out  = (float*)d_out;
  float* attn = out + (size_t)B_ * S_ * DK_;

  size_t off = 0;
  auto take = [&](size_t bytes) -> void* {
    void* p = (char*)d_ws + off;
    off += (bytes + 255) / 256 * 256;
    return p;
  };
  float*  Ts    = (float*)take((size_t)B_ * DK_ * DK_ * 4);
  float*  bqp   = (float*)take((size_t)B_ * DK_ * 4);
  bf16_t* WqpT  = (bf16_t*)take((size_t)B_ * DK_ * D_ * 2);
  bf16_t* WkT   = (bf16_t*)take((size_t)DK_ * D_ * 2);
  bf16_t* WvT   = (bf16_t*)take((size_t)DK_ * D_ * 2);
  bf16_t* xb    = (bf16_t*)take((size_t)B_ * S_ * D_ * 2);
  bf16_t* QTp   = (bf16_t*)take((size_t)B_ * S_ * DK_ * 2);
  bf16_t* Kbp   = (bf16_t*)take((size_t)B_ * S_ * DK_ * 2);
  bf16_t* Vbp   = (bf16_t*)take((size_t)B_ * S_ * DK_ * 2);
  bf16_t* Vtp   = (bf16_t*)take((size_t)B_ * DK_ * S_ * 2);
  float*  stats = (float*)take((size_t)B_ * S_ * 2 * 4);

  cvt_x_kernel<<<(B_ * S_ * D_) / (256 * 4), 256, 0, stream>>>(x, xb);
  prep_kernel<<<B_, 128, 0, stream>>>(tpnts, Wt, bt, bq, Ts, bqp);
  wqp_kernel<<<(B_ * DK_ * D_) / 256, 256, 0, stream>>>(Wq, Ts, WqpT);
  wtc_kernel<<<(DK_ * D_) / 256, 256, 0, stream>>>(Wk, WkT);
  wtc_kernel<<<(DK_ * D_) / 256, 256, 0, stream>>>(Wv, WvT);
  proj_kernel<<<dim3(S_ / 128, 3, B_), 256, 0, stream>>>(xb, WqpT, WkT, WvT, bqp, bk, bv,
                                                         QTp, Kbp, Vbp);
  vtrans_kernel<<<dim3(S_ / 32, DK_ / 32, B_), 256, 0, stream>>>(Vbp, Vtp);
  attn_kernel<<<dim3(S_ / 64, B_), 128, 0, stream>>>(QTp, Kbp, Vtp, out, attn, stats);
  norm_kernel<<<(size_t)B_ * S_ * S_ / (256 * 4), 256, 0, stream>>>(attn, stats);
}
